// MultiHeadAttention_3307124818573
// MI455X (gfx1250) — matrix operand order repaired and checked by reading, not compiled, not run on hardware
//
#include <hip/hip_runtime.h>

// ---- CDNA5 WMMA types ----
typedef __attribute__((ext_vector_type(16))) _Float16 v16h;
typedef __attribute__((ext_vector_type(8)))  _Float16 v8h;
typedef __attribute__((ext_vector_type(2)))  _Float16 v2h;
typedef __attribute__((ext_vector_type(8)))  float    v8f;
typedef __attribute__((ext_vector_type(4)))  int      v4i;

#define D_MODEL 1024
#define T_SEQ   2048
#define NBATCH  2
#define NHEAD   16
#define DHEAD   64

// ---- CDNA5 async global->LDS staging (ASYNCcnt path) ----
#if defined(__has_builtin)
#if __has_builtin(__builtin_amdgcn_global_load_async_to_lds_b128) && \
    __has_builtin(__builtin_amdgcn_s_wait_asynccnt)
#define USE_ASYNC_LDS 1
#endif
#endif
#ifndef USE_ASYNC_LDS
#define USE_ASYNC_LDS 0
#endif

typedef __attribute__((address_space(1))) v4i* gp128_t;  // global b128 ptr
typedef __attribute__((address_space(3))) v4i* lp128_t;  // LDS b128 ptr

__device__ __forceinline__ void stage_b128(const _Float16* g, _Float16* l) {
#if USE_ASYNC_LDS
  __builtin_amdgcn_global_load_async_to_lds_b128((gp128_t)g, (lp128_t)l, 0, 0);
#else
  *(v8h*)l = *(const v8h*)g;
#endif
}
// wait until at most N of this wave's async loads remain in flight
template <int N>
__device__ __forceinline__ void stage_wait() {
#if USE_ASYNC_LDS
  __builtin_amdgcn_s_wait_asynccnt(N);
#endif
}

union AFrag { v16h f; v2h p[8]; };

// A-matrix 16x32 f16 layout: lane l (M=l&15); VGPR v holds K pair.
__device__ __forceinline__ int a_kidx(int v, int hi) {
  int base = hi ? 8 : 0;
  return (v < 4) ? (base + 2 * v) : (16 + base + 2 * (v - 4));
}
// B-matrix 32x16 f16 layout: lane l (N=l&15); VGPR v holds K pair, same K order as A.
__device__ __forceinline__ int b_kidx(int v, int hi) { return a_kidx(v, hi); }

__device__ __forceinline__ v8f wmma_f16(const AFrag& a, const AFrag& b, v8f c) {
  return __builtin_amdgcn_wmma_f32_16x16x32_f16(false, a.f, false, b.f,
                                                (short)0, c, false, false);
}

__device__ __forceinline__ float rmax16(float x) {
  #pragma unroll
  for (int m = 8; m >= 1; m >>= 1) x = fmaxf(x, __shfl_xor(x, m, 16));
  return x;
}
__device__ __forceinline__ float rsum16(float x) {
  #pragma unroll
  for (int m = 8; m >= 1; m >>= 1) x += __shfl_xor(x, m, 16);
  return x;
}

// ---------------------------------------------------------------------------
// Kernel 0: one-shot f32 -> f16 conversion (x and all weights), vectorized.
// ---------------------------------------------------------------------------
__global__ __launch_bounds__(256) void cvt_f32_f16(const float* __restrict__ in,
                                                   _Float16* __restrict__ out,
                                                   int n) {
  const int i = (blockIdx.x * 256 + threadIdx.x) * 8;
  if (i < n) {
    const float4 a = *(const float4*)(in + i);
    const float4 b = *(const float4*)(in + i + 4);
    v8h o;
    o[0] = (_Float16)a.x; o[1] = (_Float16)a.y;
    o[2] = (_Float16)a.z; o[3] = (_Float16)a.w;
    o[4] = (_Float16)b.x; o[5] = (_Float16)b.y;
    o[6] = (_Float16)b.z; o[7] = (_Float16)b.w;
    *(v8h*)(out + i) = o;
  }
}

// ---------------------------------------------------------------------------
// Kernel 1: QKV projection, pure-f16 GEMM.  Y = xh @ Wh^T.
// Block tile 128x64, K-step 32, double-buffered async LDS staging.
// 4 waves x (32 rows x 64 cols) = 8 WMMA/step/wave.
// grid = (BT/128, D_OUT/64, 3); z: 0->Q, 1->K (row-major), 2->V^T [b][h][d][t].
// ---------------------------------------------------------------------------
__global__ __launch_bounds__(128) void qkv_gemm(
    const _Float16* __restrict__ xh, const _Float16* __restrict__ Wqh,
    const _Float16* __restrict__ Wkh, const _Float16* __restrict__ Wvh,
    _Float16* __restrict__ Qo, _Float16* __restrict__ Ko,
    _Float16* __restrict__ Vt)
{
  __shared__ _Float16 As[2][128 * 32];   // 2 x 8 KB
  __shared__ _Float16 Bs[2][64 * 32];    // 2 x 4 KB
  const int tid  = threadIdx.x;
  const int wave = tid >> 5;
  const int lane = tid & 31;
  const int hi   = lane >> 4;
  const int ln   = lane & 15;
  const int mbase = blockIdx.x * 128;
  const int ebase = blockIdx.y * 64;
  const int wsel  = blockIdx.z;
  const _Float16* W = (wsel == 0) ? Wqh : ((wsel == 1) ? Wkh : Wvh);

  v8f acc[2][4] = {};
  const int brow  = tid & 63;          // B-tile row
  const int bpart = tid >> 6;          // 0..1 -> 16-col half of B row
  const _Float16* gA = xh + (size_t)(mbase + tid) * D_MODEL;
  const _Float16* gB = W + (size_t)(ebase + brow) * D_MODEL + bpart * 16;

  auto stage = [&](int buf, int k0) {
    #pragma unroll
    for (int c = 0; c < 4; ++c)
      stage_b128(gA + k0 + c * 8, &As[buf][tid * 32 + c * 8]);
    #pragma unroll
    for (int c = 0; c < 2; ++c)
      stage_b128(gB + k0 + c * 8, &Bs[buf][brow * 32 + bpart * 16 + c * 8]);
  };

  stage(0, 0);
  for (int kk = 0; kk < D_MODEL / 32; ++kk) {
    const int cur = kk & 1;
    if (kk < D_MODEL / 32 - 1) {
      stage(cur ^ 1, (kk + 1) * 32);   // issue next tile, no wait
      stage_wait<6>();                 // previous 6 loads complete
    } else {
      stage_wait<0>();
    }
    __syncthreads();

    AFrag a[2];
    #pragma unroll
    for (int mf = 0; mf < 2; ++mf)
      #pragma unroll
      for (int v = 0; v < 8; ++v)
        a[mf].p[v] = *(const v2h*)(&As[cur][(wave * 32 + mf * 16 + ln) * 32 +
                                            a_kidx(v, hi)]);
    #pragma unroll
    for (int s = 0; s < 4; ++s) {
      AFrag b;
      #pragma unroll
      for (int v = 0; v < 8; ++v)
        b.p[v] = *(const v2h*)(&Bs[cur][(s * 16 + ln) * 32 + b_kidx(v, hi)]);
      acc[0][s] = wmma_f16(a[0], b, acc[0][s]);
      acc[1][s] = wmma_f16(a[1], b, acc[1][s]);
    }
    __syncthreads();
  }

  const int m0 = mbase + wave * 32;
  if (wsel < 2) {
    _Float16* dst = (wsel == 0) ? Qo : Ko;
    #pragma unroll
    for (int mf = 0; mf < 2; ++mf)
      #pragma unroll
      for (int s = 0; s < 4; ++s)
        #pragma unroll
        for (int r = 0; r < 8; ++r)
          dst[(size_t)(m0 + mf * 16 + hi * 8 + r) * D_MODEL +
              ebase + s * 16 + ln] = (_Float16)acc[mf][s][r];
  } else {
    const int b = m0 / T_SEQ;
    #pragma unroll
    for (int mf = 0; mf < 2; ++mf) {
      const int tloc = (m0 % T_SEQ) + mf * 16 + hi * 8;  // 8 consecutive t
      #pragma unroll
      for (int s = 0; s < 4; ++s) {
        const int e = ebase + s * 16 + ln;
        const int h = e >> 6, d = e & 63;
        v8h pk;
        #pragma unroll
        for (int r = 0; r < 8; ++r) pk[r] = (_Float16)acc[mf][s][r];
        *(v8h*)(Vt + ((size_t)((b * NHEAD + h) * DHEAD + d)) * T_SEQ + tloc) = pk;
      }
    }
  }
}

// ---------------------------------------------------------------------------
// Kernel 2: causal flash attention, one wave per 16-query tile.
// grid = (T/64, NHEAD, NBATCH) ; block = 128 (4 waves, independent q-tiles).
// ---------------------------------------------------------------------------
__global__ __launch_bounds__(128) void flash_attn(
    const _Float16* __restrict__ Q, const _Float16* __restrict__ K,
    const _Float16* __restrict__ Vt, _Float16* __restrict__ Oo)
{
  __shared__ _Float16 P[4 * 16 * 32];
  const int tid = threadIdx.x, wave = tid >> 5, lane = tid & 31;
  const int hi = lane >> 4, ln = lane & 15;
  const int qb = blockIdx.x * 64 + wave * 16;
  const int h  = blockIdx.y, b = blockIdx.z;
  _Float16* pb = P + wave * 16 * 32;

  AFrag aq[2];
  #pragma unroll
  for (int hf = 0; hf < 2; ++hf)
    #pragma unroll
    for (int v = 0; v < 8; ++v)
      aq[hf].p[v] = *(const v2h*)(Q + (size_t)(b * T_SEQ + qb + ln) * D_MODEL +
                                  h * DHEAD + hf * 32 + a_kidx(v, hi));

  float mrow[8], lrow[8];
  #pragma unroll
  for (int r = 0; r < 8; ++r) { mrow[r] = -3.0e38f; lrow[r] = 0.0f; }
  v8f o[4] = {};

  const float scale = 0.125f;   // 1/sqrt(64)
  for (int j = 0; j < qb + 16; j += 32) {
    AFrag bk0[2], bk1[2];
    #pragma unroll
    for (int hf = 0; hf < 2; ++hf)
      #pragma unroll
      for (int v = 0; v < 8; ++v) {
        const int koff = hf * 32 + b_kidx(v, hi);
        bk0[hf].p[v] = *(const v2h*)(K + (size_t)(b * T_SEQ + j + ln) * D_MODEL +
                                     h * DHEAD + koff);
        bk1[hf].p[v] = *(const v2h*)(K + (size_t)(b * T_SEQ + j + 16 + ln) * D_MODEL +
                                     h * DHEAD + koff);
      }
    v8f z = {};
    v8f s0 = wmma_f16(aq[0], bk0[0], z);
    s0     = wmma_f16(aq[1], bk0[1], s0);
    v8f s1 = wmma_f16(aq[0], bk1[0], z);
    s1     = wmma_f16(aq[1], bk1[1], s1);

    float p0[8], p1[8];
    #pragma unroll
    for (int r = 0; r < 8; ++r) {
      const int qg = qb + r + hi * 8;
      float a0 = s0[r] * scale; if (j + ln > qg)      a0 = -3.0e38f;
      float a1 = s1[r] * scale; if (j + 16 + ln > qg) a1 = -3.0e38f;
      const float cm = rmax16(fmaxf(a0, a1));
      const float mn = fmaxf(mrow[r], cm);
      const float alpha = __expf(mrow[r] - mn);
      mrow[r] = mn;
      p0[r] = __expf(a0 - mn);
      p1[r] = __expf(a1 - mn);
      lrow[r] = lrow[r] * alpha + rsum16(p0[r] + p1[r]);
      #pragma unroll
      for (int s = 0; s < 4; ++s) o[s][r] *= alpha;
    }

    #pragma unroll
    for (int r = 0; r < 8; ++r) {
      pb[(r + hi * 8) * 32 + ln]      = (_Float16)p0[r];
      pb[(r + hi * 8) * 32 + 16 + ln] = (_Float16)p1[r];
    }
    AFrag ap;
    #pragma unroll
    for (int v = 0; v < 8; ++v)
      ap.p[v] = *(const v2h*)(pb + ln * 32 + a_kidx(v, hi));

    #pragma unroll
    for (int s = 0; s < 4; ++s) {
      AFrag bv;
      #pragma unroll
      for (int v = 0; v < 8; ++v)
        bv.p[v] = *(const v2h*)(Vt +
                     ((size_t)((b * NHEAD + h) * DHEAD + s * 16 + ln)) * T_SEQ +
                     j + b_kidx(v, hi));
      o[s] = wmma_f16(ap, bv, o[s]);
    }
  }

  #pragma unroll
  for (int s = 0; s < 4; ++s)
    #pragma unroll
    for (int r = 0; r < 8; ++r)
      Oo[(size_t)(b * T_SEQ + qb + r + hi * 8) * D_MODEL + h * DHEAD + s * 16 + ln] =
          (_Float16)(o[s][r] / lrow[r]);
}

// ---------------------------------------------------------------------------
// Kernel 3: output projection, pure-f16 GEMM + bias, f32 out. Tile 128x64,
// double-buffered async LDS staging.
// ---------------------------------------------------------------------------
__global__ __launch_bounds__(128) void out_proj(
    const _Float16* __restrict__ Ai, const _Float16* __restrict__ Woh,
    const float* __restrict__ bo, float* __restrict__ out)
{
  __shared__ _Float16 As[2][128 * 32];
  __shared__ _Float16 Bs[2][64 * 32];
  const int tid  = threadIdx.x;
  const int wave = tid >> 5;
  const int lane = tid & 31;
  const int hi   = lane >> 4;
  const int ln   = lane & 15;
  const int mbase = blockIdx.x * 128;
  const int ebase = blockIdx.y * 64;

  v8f acc[2][4] = {};
  const int brow  = tid & 63;
  const int bpart = tid >> 6;
  const _Float16* gA = Ai + (size_t)(mbase + tid) * D_MODEL;
  const _Float16* gB = Woh + (size_t)(ebase + brow) * D_MODEL + bpart * 16;

  auto stage = [&](int buf, int k0) {
    #pragma unroll
    for (int c = 0; c < 4; ++c)
      stage_b128(gA + k0 + c * 8, &As[buf][tid * 32 + c * 8]);
    #pragma unroll
    for (int c = 0; c < 2; ++c)
      stage_b128(gB + k0 + c * 8, &Bs[buf][brow * 32 + bpart * 16 + c * 8]);
  };

  stage(0, 0);
  for (int kk = 0; kk < D_MODEL / 32; ++kk) {
    const int cur = kk & 1;
    if (kk < D_MODEL / 32 - 1) {
      stage(cur ^ 1, (kk + 1) * 32);
      stage_wait<6>();
    } else {
      stage_wait<0>();
    }
    __syncthreads();

    AFrag a[2];
    #pragma unroll
    for (int mf = 0; mf < 2; ++mf)
      #pragma unroll
      for (int v = 0; v < 8; ++v)
        a[mf].p[v] = *(const v2h*)(&As[cur][(wave * 32 + mf * 16 + ln) * 32 +
                                            a_kidx(v, hi)]);
    #pragma unroll
    for (int s = 0; s < 4; ++s) {
      AFrag b;
      #pragma unroll
      for (int v = 0; v < 8; ++v)
        b.p[v] = *(const v2h*)(&Bs[cur][(s * 16 + ln) * 32 + b_kidx(v, hi)]);
      acc[0][s] = wmma_f16(a[0], b, acc[0][s]);
      acc[1][s] = wmma_f16(a[1], b, acc[1][s]);
    }
    __syncthreads();
  }

  const int m0 = mbase + wave * 32;
  #pragma unroll
  for (int s = 0; s < 4; ++s) {
    const float bias = bo[ebase + s * 16 + ln];
    #pragma unroll
    for (int mf = 0; mf < 2; ++mf)
      #pragma unroll
      for (int r = 0; r < 8; ++r)
        out[(size_t)(m0 + mf * 16 + hi * 8 + r) * D_MODEL + ebase + s * 16 + ln] =
            acc[mf][s][r] + bias;
  }
}

// ---------------------------------------------------------------------------
extern "C" void kernel_launch(void* const* d_in, const int* in_sizes, int n_in,
                              void* d_out, int out_size, void* d_ws, size_t ws_size,
                              hipStream_t stream) {
  const float* x  = (const float*)d_in[0];
  const float* Wq = (const float*)d_in[1];
  const float* Wk = (const float*)d_in[2];
  const float* Wv = (const float*)d_in[3];
  const float* Wo = (const float*)d_in[4];
  const float* bo = (const float*)d_in[5];
  float* out = (float*)d_out;

  const size_t NX = (size_t)NBATCH * T_SEQ * D_MODEL;   // 4M elements
  const size_t NW = (size_t)D_MODEL * D_MODEL;          // 1M elements
  _Float16* xh  = (_Float16*)d_ws;
  _Float16* Wqh = xh + NX;
  _Float16* Wkh = Wqh + NW;
  _Float16* Wvh = Wkh + NW;
  _Float16* Woh = Wvh + NW;
  _Float16* Q   = Woh + NW;
  _Float16* K   = Q + NX;
  _Float16* Vt  = K + NX;
  _Float16* AO  = Vt + NX;

  cvt_f32_f16<<<(int)(NX / 2048), 256, 0, stream>>>(x, xh, (int)NX);
  cvt_f32_f16<<<(int)(NW / 2048), 256, 0, stream>>>(Wq, Wqh, (int)NW);
  cvt_f32_f16<<<(int)(NW / 2048), 256, 0, stream>>>(Wk, Wkh, (int)NW);
  cvt_f32_f16<<<(int)(NW / 2048), 256, 0, stream>>>(Wv, Wvh, (int)NW);
  cvt_f32_f16<<<(int)(NW / 2048), 256, 0, stream>>>(Wo, Woh, (int)NW);

  qkv_gemm<<<dim3((NBATCH * T_SEQ) / 128, D_MODEL / 64, 3), 128, 0, stream>>>(
      xh, Wqh, Wkh, Wvh, Q, K, Vt);
  flash_attn<<<dim3(T_SEQ / 64, NHEAD, NBATCH), 128, 0, stream>>>(Q, K, Vt, AO);
  out_proj<<<dim3((NBATCH * T_SEQ) / 128, D_MODEL / 64), 128, 0, stream>>>(
      AO, Woh, bo, out);
}
